// Attend_8211977470699
// MI455X (gfx1250) — matrix operand order repaired and checked by reading, not compiled, not run on hardware
//
#include <hip/hip_runtime.h>
#include <hip/hip_bf16.h>

// Attention: B=2, H=16, N=2048, D=128, causal, fp32 I/O, fp32 bias (1,H,N,N).
// Flash-attention with V_WMMA_F32_16X16X32_F16 (f32 accumulate).
// Prep kernels cast K -> f16 (row major) and V -> f16 transposed into d_ws so
// the hot loop does pure f16 vector loads (no per-iteration cvt, no strided V).

#define BB 2
#define HH 16
#define NN 2048
#define DD 128

typedef float    v8f  __attribute__((ext_vector_type(8)));
typedef _Float16 v16h __attribute__((ext_vector_type(16)));
typedef _Float16 v8h  __attribute__((ext_vector_type(8)));

__device__ inline v8f zero8() {
    v8f z;
#pragma unroll
    for (int u = 0; u < 8; ++u) z[u] = 0.0f;
    return z;
}

// ---------------------------------------------------------------------------
// Prep 1: cast K (fp32 row-major) -> Kh (f16 row-major). 8 floats / thread.
// ---------------------------------------------------------------------------
__global__ __launch_bounds__(256) void
k_cast_kernel(const float* __restrict__ k, void* __restrict__ khv)
{
    _Float16* kh = (_Float16*)khv;
    const size_t idx = ((size_t)blockIdx.x * 256 + threadIdx.x) * 8;
    float4 a = *(const float4*)(k + idx);
    float4 b = *(const float4*)(k + idx + 4);
    v8h o;
    o[0] = (_Float16)a.x; o[1] = (_Float16)a.y;
    o[2] = (_Float16)a.z; o[3] = (_Float16)a.w;
    o[4] = (_Float16)b.x; o[5] = (_Float16)b.y;
    o[6] = (_Float16)b.z; o[7] = (_Float16)b.w;
    *(v8h*)(kh + idx) = o;
}

// ---------------------------------------------------------------------------
// Prep 2: V (fp32 [bh][j][d]) -> Vt (f16 [bh][d][j]).  32x32 LDS-tiled
// transpose, coalesced reads and writes.
// ---------------------------------------------------------------------------
__global__ __launch_bounds__(256) void
v_transpose_kernel(const float* __restrict__ v, void* __restrict__ vtv)
{
    __shared__ float tile[32][33];
    _Float16* vt = (_Float16*)vtv;

    const int bh = blockIdx.z;
    const int j0 = blockIdx.x * 32;
    const int d0 = blockIdx.y * 32;
    const float* vp = v  + (size_t)bh * NN * DD;
    _Float16*   tp = vt + (size_t)bh * NN * DD;

    const int tx = threadIdx.x;        // 0..31
    const int ty = threadIdx.y;        // 0..7
#pragma unroll
    for (int yy = 0; yy < 4; ++yy) {
        const int row = ty + yy * 8;
        tile[row][tx] = vp[(size_t)(j0 + row) * DD + d0 + tx];
    }
    __syncthreads();
#pragma unroll
    for (int yy = 0; yy < 4; ++yy) {
        const int drow = ty + yy * 8;
        tp[(size_t)(d0 + drow) * NN + j0 + tx] = (_Float16)tile[tx][drow];
    }
}

// ---------------------------------------------------------------------------
// Main flash-attention kernel. 4 wave32 per block, 16 query rows per wave.
// ---------------------------------------------------------------------------
__global__ __launch_bounds__(128) void
attend_fwd_kernel(const float* __restrict__ q,
                  const void*  __restrict__ khv,   // f16 [bh][j][d]
                  const void*  __restrict__ vtv,   // f16 [bh][d][j]
                  const unsigned char* __restrict__ mask,
                  const float* __restrict__ bias,
                  float* __restrict__ out)
{
    const _Float16* khg = (const _Float16*)khv;
    const _Float16* vtg = (const _Float16*)vtv;

    // per-wave 16x32 f16 P staging buffer (C-layout -> A-layout transpose)
    __shared__ _Float16 p_lds[4][16][32];   // 4 KB

    const int lane = threadIdx.x & 31;
    const int wave = threadIdx.x >> 5;
    const int half = lane >> 4;
    const int l16  = lane & 15;

    const int bh = blockIdx.y;
    const int b  = bh / HH;
    const int h  = bh % HH;
    const int i0 = (blockIdx.x * 4 + wave) * 16;

    const float scale = 0.08838834764831845f;   // 1/sqrt(128)

    const float*    qp = q   + (size_t)bh * NN * DD;
    const _Float16* kp = khg + (size_t)bh * NN * DD;
    const _Float16* vp = vtg + (size_t)bh * NN * DD;
    const float*    bp = bias + (size_t)h * NN * NN;
    const unsigned char* mp = mask + (size_t)b * NN;
    float* op = out + (size_t)bh * NN * DD;

    // ---- Preload Q as 4 A-layout f16 fragments (one-time f32->f16 cast) ----
    v16h qa[4];
    {
        const float* qrow = qp + (size_t)(i0 + l16) * DD;
#pragma unroll
        for (int c = 0; c < 4; ++c) {
            const float* src = qrow + c * 32 + half * 16;
            float4 f0 = *(const float4*)(src + 0);
            float4 f1 = *(const float4*)(src + 4);
            float4 f2 = *(const float4*)(src + 8);
            float4 f3 = *(const float4*)(src + 12);
            qa[c][0]  = (_Float16)f0.x; qa[c][1]  = (_Float16)f0.y;
            qa[c][2]  = (_Float16)f0.z; qa[c][3]  = (_Float16)f0.w;
            qa[c][4]  = (_Float16)f1.x; qa[c][5]  = (_Float16)f1.y;
            qa[c][6]  = (_Float16)f1.z; qa[c][7]  = (_Float16)f1.w;
            qa[c][8]  = (_Float16)f2.x; qa[c][9]  = (_Float16)f2.y;
            qa[c][10] = (_Float16)f2.z; qa[c][11] = (_Float16)f2.w;
            qa[c][12] = (_Float16)f3.x; qa[c][13] = (_Float16)f3.y;
            qa[c][14] = (_Float16)f3.z; qa[c][15] = (_Float16)f3.w;
        }
    }

    v8f o[8];
#pragma unroll
    for (int d = 0; d < 8; ++d) o[d] = zero8();
    float m_i[8], l_i[8];
#pragma unroll
    for (int r = 0; r < 8; ++r) { m_i[r] = -3.0e38f; l_i[r] = 0.0f; }

    for (int j0 = 0; j0 < i0 + 16; j0 += 32) {
        // prefetch next bias tile (dominant HBM stream) while we compute
        const int jn = j0 + 32;
        if (jn < i0 + 16) {
#pragma unroll
            for (int r = 0; r < 8; ++r) {
                const int irow = i0 + r + 8 * half;
                __builtin_prefetch(bp + (size_t)irow * NN + jn + l16, 0, 3);
            }
        }

        // ---- S = Q K^T : two 16x16 C-tiles ----
        v8f s[2];
#pragma unroll
        for (int t = 0; t < 2; ++t) {
            v8f acc = zero8();
            // B-layout: lane col n = key (j0+t*16+l16); lanes<16 k-rows d 0..15
            // of each 32-chunk, lanes>=16 k-rows 16..31.  f16, contiguous.
            const _Float16* krow =
                kp + (size_t)(j0 + t * 16 + l16) * DD + half * 16;
#pragma unroll
            for (int c = 0; c < 4; ++c) {
                v16h bf = *(const v16h*)(krow + c * 32);
                acc = __builtin_amdgcn_wmma_f32_16x16x32_f16(
                        false, qa[c], false, bf, (short)0, acc, false, false);
            }
            s[t] = acc;
        }

        // ---- scale + bias + causal/key mask in C-layout registers ----
#pragma unroll
        for (int t = 0; t < 2; ++t) {
            const int jcol = j0 + t * 16 + l16;
            const bool kmask = (mp[jcol] != 0);
#pragma unroll
            for (int r = 0; r < 8; ++r) {
                const int irow = i0 + r + 8 * half;
                float val = s[t][r] * scale + bp[(size_t)irow * NN + jcol];
                if ((jcol > irow) || !kmask) val = -3.0e38f;
                s[t][r] = val;
            }
        }

        // ---- online softmax (16-lane shfl_xor row reductions) ----
        float alpha[8];
#pragma unroll
        for (int r = 0; r < 8; ++r) {
            float mt = fmaxf(s[0][r], s[1][r]);
#pragma unroll
            for (int off = 1; off < 16; off <<= 1)
                mt = fmaxf(mt, __shfl_xor(mt, off, 32));
            const float mnew = fmaxf(m_i[r], mt);
            alpha[r] = __expf(m_i[r] - mnew);
            m_i[r] = mnew;
            float p0 = __expf(s[0][r] - mnew);
            float p1 = __expf(s[1][r] - mnew);
            s[0][r] = p0; s[1][r] = p1;
            float rs = p0 + p1;
#pragma unroll
            for (int off = 1; off < 16; off <<= 1)
                rs += __shfl_xor(rs, off, 32);
            l_i[r] = l_i[r] * alpha[r] + rs;
        }

#pragma unroll
        for (int d = 0; d < 8; ++d)
#pragma unroll
            for (int r = 0; r < 8; ++r) o[d][r] *= alpha[r];

        // ---- P through LDS: scatter C-layout f16, gather A-layout 16B ----
        _Float16 (*pl)[32] = p_lds[wave];
#pragma unroll
        for (int t = 0; t < 2; ++t) {
            const int jc = t * 16 + l16;
#pragma unroll
            for (int r = 0; r < 8; ++r)
                pl[r + 8 * half][jc] = (_Float16)s[t][r];
        }
        __builtin_amdgcn_wave_barrier();
        v16h pa;
        {
            const _Float16* row = &pl[l16][half * 16];
            v8h lo = *(const v8h*)(row);
            v8h hi = *(const v8h*)(row + 8);
#pragma unroll
            for (int u = 0; u < 8; ++u) { pa[u] = lo[u]; pa[8 + u] = hi[u]; }
        }

        // ---- O += P V : one contiguous v16h load of Vt per D-chunk ----
#pragma unroll
        for (int d = 0; d < 8; ++d) {
            // B-layout: lane col n = d*16+l16 (a row of Vt), k-rows are the
            // 16 consecutive keys j0+half*16.. -> contiguous f16 in Vt.
            v16h vb = *(const v16h*)
                (vp + (size_t)(d * 16 + l16) * NN + j0 + half * 16);
            o[d] = __builtin_amdgcn_wmma_f32_16x16x32_f16(
                       false, pa, false, vb, (short)0, o[d], false, false);
        }
    }

    // ---- normalize and store ----
#pragma unroll
    for (int d = 0; d < 8; ++d) {
#pragma unroll
        for (int r = 0; r < 8; ++r) {
            const int irow = i0 + r + 8 * half;
            op[(size_t)irow * DD + d * 16 + l16] = o[d][r] / l_i[r];
        }
    }
}

extern "C" void kernel_launch(void* const* d_in, const int* in_sizes, int n_in,
                              void* d_out, int out_size, void* d_ws, size_t ws_size,
                              hipStream_t stream) {
    const float*         q    = (const float*)d_in[0];
    const float*         k    = (const float*)d_in[1];
    const float*         v    = (const float*)d_in[2];
    const unsigned char* mask = (const unsigned char*)d_in[3];
    const float*         bias = (const float*)d_in[4];
    float*               out  = (float*)d_out;

    const size_t elems    = (size_t)BB * HH * NN * DD;   // 8,388,608
    const size_t kh_bytes = elems * 2;                   // 16 MB
    void* kh = d_ws;
    void* vt = (void*)((char*)d_ws + kh_bytes);          // needs 32 MB total

    // Prep: K -> f16, V -> f16 transposed (one-time cast amortized over N^2)
    k_cast_kernel<<<dim3((unsigned)(elems / (256 * 8))), dim3(256), 0, stream>>>(k, kh);
    v_transpose_kernel<<<dim3(NN / 32, DD / 32, BB * HH), dim3(32, 8), 0, stream>>>(v, vt);

    // Flash attention
    dim3 grid(NN / 64, BB * HH);
    dim3 block(128);
    attend_fwd_kernel<<<grid, block, 0, stream>>>(q, kh, vt, mask, bias, out);
}
